// NGPRadianceFieldSingleMlp_85435489452144
// MI455X (gfx1250) — matrix operand order repaired and checked by reading, not compiled, not run on hardware
//
#include <hip/hip_runtime.h>
#include <hip/hip_bf16.h>
#include <math.h>

// ---------------------------------------------------------------------------
// Fused Instant-NGP radiance field for gfx1250 (MI455X):
//   hash-grid encode (L2-resident tables) + SH  ->  LDS (f16)
//   4-layer MLP via v_wmma_f32_16x16x32_f16, activations stay in LDS
// Block: 256 threads = 8 wave32; each wave owns a 16-sample WMMA tile.
// ---------------------------------------------------------------------------

typedef _Float16 h16;
typedef __attribute__((ext_vector_type(16))) _Float16 v16h;
typedef __attribute__((ext_vector_type(8)))  _Float16 v8h;
typedef __attribute__((ext_vector_type(8)))  float    v8f;

#define HASH_T   (1u << 19)          // 2^19 entries per level
#define LDS_STR  72                  // row stride in halves (bank-conflict pad)

__constant__ float RES_TAB[16] = {
    16.f, 23.f, 33.f, 48.f, 70.f, 101.f, 147.f, 212.f,
    307.f, 445.f, 645.f, 933.f, 1351.f, 1955.f, 2830.f, 4096.f};

union V16U { v16h v; v8h h[2]; };

// One 16xN GEMM tile, K = 64 (two 16x16x32 WMMA k-steps), NT 16-wide N tiles.
template <int NT>
__device__ __forceinline__ void gemm_tile_k64(const h16* __restrict__ aTile,
                                              const h16* __restrict__ wt,
                                              int lane, v8f acc[NT]) {
    const int col = lane & 15;      // A row / B col / D col
    const int hi  = lane >> 4;      // which K-slice this half-wave holds
    const v8f zero = {0.f, 0.f, 0.f, 0.f, 0.f, 0.f, 0.f, 0.f};
#pragma unroll
    for (int nt = 0; nt < NT; ++nt) acc[nt] = zero;

#pragma unroll
    for (int ks = 0; ks < 2; ++ks) {
        const int kb = ks * 32;
        // A fragment (16x32 f16): lanes<16 hold K kb+0..7 / kb+16..23,
        //                         lanes>=16 hold K kb+8..15 / kb+24..31.
        V16U a;
        const h16* ap = aTile + col * LDS_STR + kb + hi * 8;
        a.h[0] = *(const v8h*)(ap);
        a.h[1] = *(const v8h*)(ap + 16);
#pragma unroll
        for (int nt = 0; nt < NT; ++nt) {
            // B fragment (32x16 f16): lanes<16 K kb+0..7 / kb+16..23, lanes>=16 K kb+8..15 / kb+24..31.
            const h16* bp = wt + (nt * 16 + col) * LDS_STR + kb + hi * 8;
            V16U b;
            b.h[0] = *(const v8h*)(bp);
            b.h[1] = *(const v8h*)(bp + 16);
            acc[nt] = __builtin_amdgcn_wmma_f32_16x16x32_f16(
                false, a.v, false, b.v, (short)0, acc[nt], false, false);
        }
    }
}

template <int NT>
__device__ __forceinline__ void relu_store(const v8f acc[NT],
                                           h16* __restrict__ outTile, int lane) {
    const int col = lane & 15;
    const int hi  = lane >> 4;
#pragma unroll
    for (int nt = 0; nt < NT; ++nt) {
#pragma unroll
        for (int r = 0; r < 8; ++r) {
            float v = acc[nt][r];
            v = v > 0.f ? v : 0.f;
            outTile[(r + hi * 8) * LDS_STR + nt * 16 + col] = (h16)v;
        }
    }
}

__global__ __launch_bounds__(256) void ngp_fused_kernel(
    const float* __restrict__ pos, const float* __restrict__ dirs,
    const float* __restrict__ aabb, const float* __restrict__ tables,
    const float* __restrict__ W0, const float* __restrict__ W1,
    const float* __restrict__ W2, const float* __restrict__ W3,
    float* __restrict__ out, int n) {
    // f16 weights, transposed (N-major) so B fragments are contiguous b128 loads
    __shared__ alignas(16) h16 wt0[64 * LDS_STR];
    __shared__ alignas(16) h16 wt1[64 * LDS_STR];
    __shared__ alignas(16) h16 wt2[64 * LDS_STR];
    __shared__ alignas(16) h16 wt3[16 * LDS_STR];
    __shared__ alignas(16) h16 actA[8 * 16 * LDS_STR];
    __shared__ alignas(16) h16 actB[8 * 16 * LDS_STR];
    __shared__ float sel[128];

    const int t = threadIdx.x;

    // ---- phase 0: weights -> LDS (transpose, f16, zero-pad) ----------------
    for (int i = t; i < 64 * 64; i += 256) {
        const int nn = i & 63, k = i >> 6;
        wt0[nn * LDS_STR + k] = (h16)((k < 48) ? W0[k * 64 + nn] : 0.f);
        wt1[nn * LDS_STR + k] = (h16)W1[k * 64 + nn];
        wt2[nn * LDS_STR + k] = (h16)W2[k * 64 + nn];
    }
    for (int i = t; i < 16 * 64; i += 256) {
        const int nn = i & 15, k = i >> 4;
        wt3[nn * LDS_STR + k] = (h16)((nn < 4) ? W3[k * 4 + nn] : 0.f);
    }

    // ---- phase 1: hash-grid + SH encoding -> actA --------------------------
    // 2 threads per sample: thread t handles sample si = t>>1, levels half*8..+8
    const int si = t >> 1;
    const int half = t & 1;
    const long s = (long)blockIdx.x * 128 + si;
    h16* myact = &actA[si * LDS_STR];

    const float a0 = aabb[0], a1 = aabb[1], a2 = aabb[2];
    const float inv0 = 1.f / (aabb[3] - a0);
    const float inv1 = 1.f / (aabb[4] - a1);
    const float inv2 = 1.f / (aabb[5] - a2);

    if (s < n) {
        const float xn0 = (pos[s * 3 + 0] - a0) * inv0;
        const float xn1 = (pos[s * 3 + 1] - a1) * inv1;
        const float xn2 = (pos[s * 3 + 2] - a2) * inv2;

        if (half == 0) {
            const bool inb = xn0 > 0.f && xn0 < 1.f && xn1 > 0.f && xn1 < 1.f &&
                             xn2 > 0.f && xn2 < 1.f;
            sel[si] = inb ? 1.f : 0.f;

            const float x = dirs[s * 3 + 0];
            const float y = dirs[s * 3 + 1];
            const float z = dirs[s * 3 + 2];
            const float xy = x * y, xz = x * z, yz = y * z;
            const float x2 = x * x, y2 = y * y, z2 = z * z;
            float sh[16];
            sh[0]  = 0.28209479177387814f;
            sh[1]  = -0.48860251190291987f * y;
            sh[2]  =  0.48860251190291987f * z;
            sh[3]  = -0.48860251190291987f * x;
            sh[4]  =  1.0925484305920792f * xy;
            sh[5]  = -1.0925484305920792f * yz;
            sh[6]  =  0.94617469575756f * z2 - 0.31539156525252f;
            sh[7]  = -1.0925484305920792f * xz;
            sh[8]  =  0.5462742152960396f * (x2 - y2);
            sh[9]  =  0.5900435899266435f * y * (-3.f * x2 + y2);
            sh[10] =  2.890611442640554f * xy * z;
            sh[11] =  0.4570457994644657f * y * (1.f - 5.f * z2);
            sh[12] =  0.3731763325901154f * z * (5.f * z2 - 3.f);
            sh[13] =  0.4570457994644657f * x * (1.f - 5.f * z2);
            sh[14] =  1.445305721320277f * z * (x2 - y2);
            sh[15] =  0.5900435899266435f * x * (-x2 + 3.f * y2);
#pragma unroll
            for (int j = 0; j < 16; ++j) myact[32 + j] = (h16)sh[j];
#pragma unroll
            for (int j = 48; j < 64; ++j) myact[j] = (h16)0.f;  // K pad 48->64
        }

        for (int l = 0; l < 8; ++l) {
            const int lvl = half * 8 + l;
            const float res = RES_TAB[lvl];
            const float sx = xn0 * res, sy = xn1 * res, sz = xn2 * res;
            const float bx = floorf(sx), by = floorf(sy), bz = floorf(sz);
            const float fx = sx - bx, fy = sy - by, fz = sz - bz;
            const unsigned ux = (unsigned)(int)bx;
            const unsigned uy = (unsigned)(int)by;
            const unsigned uz = (unsigned)(int)bz;
            float f0 = 0.f, f1 = 0.f;
#pragma unroll
            for (int c = 0; c < 8; ++c) {
                const unsigned ox = (c >> 2) & 1, oy = (c >> 1) & 1, oz = c & 1;
                const unsigned hsh = (ux + ox) * 1u ^ (uy + oy) * 2654435761u ^
                                     (uz + oz) * 805459861u;
                const unsigned idx = hsh & (HASH_T - 1u);
                const float* fp = tables + ((size_t)lvl * HASH_T + idx) * 2;
                const float wgt = (ox ? fx : 1.f - fx) * (oy ? fy : 1.f - fy) *
                                  (oz ? fz : 1.f - fz);
                f0 += wgt * fp[0];
                f1 += wgt * fp[1];
            }
            myact[lvl * 2 + 0] = (h16)f0;
            myact[lvl * 2 + 1] = (h16)f1;
        }
    } else {
        if (half == 0) {
            sel[si] = 0.f;
            for (int j = 32; j < 64; ++j) myact[j] = (h16)0.f;
        }
        for (int l = 0; l < 8; ++l) {
            const int lvl = half * 8 + l;
            myact[lvl * 2 + 0] = (h16)0.f;
            myact[lvl * 2 + 1] = (h16)0.f;
        }
    }
    __syncthreads();

    // ---- phase 2: MLP, one 16-sample tile per wave -------------------------
    const int wv = t >> 5;
    const int lane = t & 31;
    h16* aT = &actA[wv * 16 * LDS_STR];
    h16* bT = &actB[wv * 16 * LDS_STR];

    {   // layer 0: (padded 64) -> 64, ReLU
        v8f acc[4];
        gemm_tile_k64<4>(aT, wt0, lane, acc);
        relu_store<4>(acc, bT, lane);
    }
    __syncthreads();
    {   // layer 1: 64 -> 64, ReLU
        v8f acc[4];
        gemm_tile_k64<4>(bT, wt1, lane, acc);
        relu_store<4>(acc, aT, lane);
    }
    __syncthreads();
    {   // layer 2: 64 -> 64, ReLU
        v8f acc[4];
        gemm_tile_k64<4>(aT, wt2, lane, acc);
        relu_store<4>(acc, bT, lane);
    }
    __syncthreads();
    {   // layer 3: 64 -> 4 (padded to 16), heads
        v8f acc[1];
        gemm_tile_k64<1>(bT, wt3, lane, acc);
        const int col = lane & 15;
        const int hi  = lane >> 4;
        if (col < 4) {
#pragma unroll
            for (int r = 0; r < 8; ++r) {
                const int rr = r + hi * 8;
                const long sg = (long)blockIdx.x * 128 + wv * 16 + rr;
                if (sg < n) {
                    const float v = acc[0][r];
                    if (col < 3) {
                        out[sg * 3 + col] = 1.f / (1.f + expf(-v));       // rgb
                    } else {
                        out[(long)n * 3 + sg] =
                            expf(v - 1.f) * sel[wv * 16 + rr];            // density
                    }
                }
            }
        }
    }
}

extern "C" void kernel_launch(void* const* d_in, const int* in_sizes, int n_in,
                              void* d_out, int out_size, void* d_ws, size_t ws_size,
                              hipStream_t stream) {
    (void)n_in; (void)out_size; (void)d_ws; (void)ws_size;
    const float* pos    = (const float*)d_in[0];
    const float* dirs   = (const float*)d_in[1];
    const float* aabb   = (const float*)d_in[2];
    const float* tables = (const float*)d_in[3];
    const float* W0     = (const float*)d_in[4];
    const float* W1     = (const float*)d_in[5];
    const float* W2     = (const float*)d_in[6];
    const float* W3     = (const float*)d_in[7];
    float* out = (float*)d_out;

    const int n = in_sizes[0] / 3;           // N samples
    const int blocks = (n + 127) / 128;      // 128 samples per block

    hipLaunchKernelGGL(ngp_fused_kernel, dim3(blocks), dim3(256), 0, stream,
                       pos, dirs, aabb, tables, W0, W1, W2, W3, out, n);
}
